// WaveGlow_26353919328297
// MI455X (gfx1250) — compile-verified, matrix operand order checked by reading, not run on hardware
//
#include <hip/hip_runtime.h>
#include <cstddef>
#include <cstdint>

// ---------------------------------------------------------------------------
// WaveGlow inference for MI455X (gfx1250, wave32).
// Conv-as-GEMM via v_wmma_f32_16x16x32_bf16; LDS double-buffered staging via
// global_load_async_to_lds_b128 (ASYNCcnt).  Activations time-major [T][C];
// weights pre-converted to bf16 per flow so staging is pure 16B copies.
// ---------------------------------------------------------------------------

typedef __attribute__((ext_vector_type(8)))  float  v8f;
typedef __attribute__((ext_vector_type(16))) __bf16 v16bf;
typedef int v4i __attribute__((vector_size(16)));
typedef __attribute__((address_space(1))) v4i* global_v4i;
typedef __attribute__((address_space(3))) v4i* lds_v4i;

#define T_AUDIO 32768
#define TG      4096
#define NCH     256
#define NMEL    80
#define SMEL    640
#define CONDCH  4096
#define KUP     1024
#define SUP     256
#define TMEL    128

#if defined(__has_builtin)
#  if __has_builtin(__builtin_amdgcn_global_load_async_to_lds_b128)
#    define USE_ASYNC 1
#  endif
#endif
#ifndef USE_ASYNC
#  define USE_ASYNC 0
#endif

// flat address == global address on AMDGPU, so inttoptr is safe for AS(1);
// LDS must use a real addrspacecast (C-style cast), never an integer round-trip.
#define AS_GLOBAL(p) ((global_v4i)(uintptr_t)(p))
#define AS_LDS(p)    ((lds_v4i)(p))

__device__ __forceinline__ void wait_async0() {
#if USE_ASYNC
#  if __has_builtin(__builtin_amdgcn_s_wait_asynccnt)
  __builtin_amdgcn_s_wait_asynccnt(0);
#  else
  asm volatile("s_wait_asynccnt 0x0" ::: "memory");
#  endif
#endif
}

__device__ __forceinline__ unsigned short f2bf(float x) {
  unsigned u = __float_as_uint(x);
  unsigned r = u + 0x7FFFu + ((u >> 16) & 1u);   // round-to-nearest-even
  return (unsigned short)(r >> 16);
}
__device__ __forceinline__ float bf2f(unsigned short h) {
  return __uint_as_float(((unsigned)h) << 16);
}

// -------------------------- weight conversion ------------------------------
__global__ void cvt_w_kernel(const float* __restrict__ src, unsigned short* __restrict__ dst, int n) {
  int idx = blockIdx.x * blockDim.x + threadIdx.x;
  if (idx < n) dst[idx] = f2bf(src[idx]);
}
// in_w [512][256][3] -> [512][768] with kk = tap*256 + ci (contiguous K)
__global__ void cvt_convw_kernel(const float* __restrict__ src, unsigned short* __restrict__ dst) {
  int idx = blockIdx.x * blockDim.x + threadIdx.x;
  if (idx >= 512 * 768) return;
  int o = idx / 768, r = idx - o * 768;
  int tap = r >> 8, ci = r & 255;
  dst[idx] = f2bf(src[o * 768 + ci * 3 + tap]);
}

// -------------------------- upsampler (transposed conv) --------------------
__global__ void upsample_kernel(const float* __restrict__ spect,
                                const float* __restrict__ up_w,
                                const float* __restrict__ up_b,
                                unsigned short* __restrict__ s_bf) {
  int idx = blockIdx.x * blockDim.x + threadIdx.x;
  if (idx >= NMEL * T_AUDIO) return;
  int t = idx & (T_AUDIO - 1);
  int o = idx >> 15;
  float acc = up_b[o];
  int j = t >> 8;
  int k = t - (j << 8);
  for (; j >= 0 && k < KUP; --j, k += SUP) {
    #pragma unroll 4
    for (int i = 0; i < NMEL; ++i)
      acc += spect[i * TMEL + j] * up_w[((size_t)(i * NMEL + o)) * KUP + k];
  }
  s_bf[(size_t)(t >> 3) * SMEL + o * 8 + (t & 7)] = f2bf(acc);
}

// -------------------------- WMMA fragment loader ----------------------------
// LDS tile rows padded to 40 bf16 (20 dwords).  CDNA5 16-bit A layout:
// lane<16 -> row lane, K {0..7,16..23}; lane>=16 -> row lane-16, K {8..15,24..31}.
// B mirrors with lane = column; both read the same dword pattern.
__device__ __forceinline__ v16bf load_frag(const unsigned* lds, int rowBase, int lane) {
  int r = rowBase + (lane & 15);
  const unsigned* p = lds + r * 20 + ((lane >> 4) << 2);
  union { unsigned u[8]; v16bf v; } f;
  #pragma unroll
  for (int j = 0; j < 4; ++j) { f.u[j] = p[j]; f.u[4 + j] = p[8 + j]; }
  return f.v;
}

// -------------------------- generic WMMA GEMM ------------------------------
// Y[t][o] = bias[o] + sum_{tap,ci} X[t+(tap-1)*dil][ci] * Wb[o][tap*CI+ci]
// X: bf16 [4096][CI]; Wb: bf16 [O][CI*TAPS]; Y: f32 or bf16 [4096][O].
// 128(t) x 128(o) tile, 8 wave32s, K-step 32, double-buffered LDS.
template<int CI, int TAPS, int BF16OUT>
__global__ __launch_bounds__(256)
void gemm_wmma_kernel(const unsigned short* __restrict__ Wb,
                      const float* __restrict__ bias,
                      const unsigned short* __restrict__ X,
                      float* __restrict__ Yf, unsigned short* __restrict__ Ybf,
                      int O, int dil, const unsigned short* __restrict__ zeros) {
  constexpr int KTOT = CI * TAPS;
  constexpr int NK   = KTOT / 32;
  __shared__ unsigned As[2][128 * 20];
  __shared__ unsigned Ws[2][128 * 20];
  const int tid   = threadIdx.x;
  const int lane  = tid & 31, wid = tid >> 5;
  const int waveT = wid >> 1, waveO = wid & 1;       // 4 x 2 wave grid
  const int tBlock = blockIdx.x * 128, oBlock = blockIdx.y * 128;
  const int row2 = tid >> 1, half = tid & 1;

  // stage K-chunk `ks` (0..NK-1) into LDS buffer `buf`
  auto stage = [&](int ks, int buf) {
    const int kb = ks * 32;
    int ci0, shift;
    if constexpr (TAPS == 1) { ci0 = kb; shift = 0; }
    else { int tap = kb / CI; ci0 = kb - tap * CI; shift = (tap - 1) * dil; }

    // ---- A tile: 128 t-rows x 32 k (zero-pad shifted rows via redirect)
    const unsigned short* ga;
    {
      int tsrc = tBlock + row2 + shift;
      bool oob = (TAPS == 3) && ((unsigned)tsrc >= (unsigned)TG);
      ga = oob ? zeros : (X + (size_t)tsrc * CI + ci0 + half * 16);
    }
    void* la = (void*)&As[buf][row2 * 20 + half * 8];
    // ---- W tile: 128 o-rows x 32 k (bf16, K contiguous for both layouts)
    const unsigned short* gw = Wb + (size_t)(oBlock + row2) * KTOT + kb + half * 16;
    void* lw = (void*)&Ws[buf][row2 * 20 + half * 8];
#if USE_ASYNC
    __builtin_amdgcn_global_load_async_to_lds_b128(AS_GLOBAL(ga), AS_LDS(la), 0, 0);
    __builtin_amdgcn_global_load_async_to_lds_b128(AS_GLOBAL(ga), AS_LDS(la), 16, 0);
    __builtin_amdgcn_global_load_async_to_lds_b128(AS_GLOBAL(gw), AS_LDS(lw), 0, 0);
    __builtin_amdgcn_global_load_async_to_lds_b128(AS_GLOBAL(gw), AS_LDS(lw), 16, 0);
#else
    uint4 a0 = ((const uint4*)ga)[0], a1 = ((const uint4*)ga)[1];
    uint4 w0 = ((const uint4*)gw)[0], w1 = ((const uint4*)gw)[1];
    ((uint4*)la)[0] = a0; ((uint4*)la)[1] = a1;
    ((uint4*)lw)[0] = w0; ((uint4*)lw)[1] = w1;
#endif
  };

  v8f acc[2][4];
  #pragma unroll
  for (int mi = 0; mi < 2; ++mi)
    #pragma unroll
    for (int ni = 0; ni < 4; ++ni)
      #pragma unroll
      for (int r = 0; r < 8; ++r) acc[mi][ni][r] = 0.f;

  stage(0, 0);
  wait_async0();
  __syncthreads();

  for (int k = 0; k < NK; ++k) {
    const int buf = k & 1;
    if (k + 1 < NK) stage(k + 1, buf ^ 1);     // overlap staging with WMMA

    v16bf a[2], b[4];
    #pragma unroll
    for (int mi = 0; mi < 2; ++mi) a[mi] = load_frag(As[buf], waveT * 32 + mi * 16, lane);
    #pragma unroll
    for (int ni = 0; ni < 4; ++ni) b[ni] = load_frag(Ws[buf], waveO * 64 + ni * 16, lane);
    #pragma unroll
    for (int mi = 0; mi < 2; ++mi)
      #pragma unroll
      for (int ni = 0; ni < 4; ++ni)
        acc[mi][ni] = __builtin_amdgcn_wmma_f32_16x16x32_bf16(
            false, a[mi], false, b[ni], (short)0, acc[mi][ni], false, false);

    wait_async0();
    __syncthreads();
  }

  // ---- epilogue: D layout VGPR r -> t row r (+8 hi lanes), col = lane&15
  #pragma unroll
  for (int mi = 0; mi < 2; ++mi)
    #pragma unroll
    for (int ni = 0; ni < 4; ++ni) {
      int tRow = tBlock + waveT * 32 + mi * 16 + ((lane >> 4) << 3);
      int oCol = oBlock + waveO * 64 + ni * 16 + (lane & 15);
      float bv = bias[oCol];
      #pragma unroll
      for (int r = 0; r < 8; ++r) {
        float y = acc[mi][ni][r] + bv;
        size_t off = (size_t)(tRow + r) * O + oCol;
        if constexpr (BF16OUT) Ybf[off] = f2bf(y);
        else                   Yf[off]  = y;
      }
    }
}

// -------------------------- small scalar kernels ---------------------------
__global__ void start_kernel(const float* __restrict__ audio,
                             const float* __restrict__ w, const float* __restrict__ b,
                             float* __restrict__ xf, unsigned short* __restrict__ xbf,
                             int off, int n_half) {
  int idx = blockIdx.x * blockDim.x + threadIdx.x;
  if (idx >= TG * NCH) return;
  int o = idx & 255, t = idx >> 8;
  float acc = b[o];
  for (int ci = 0; ci < n_half; ++ci)
    acc += audio[(size_t)(off + ci) * TG + t] * w[o * n_half + ci];
  xf [(size_t)t * NCH + o] = acc;
  xbf[(size_t)t * NCH + o] = f2bf(acc);
}

__global__ void gate_kernel(const float* __restrict__ xin,
                            const unsigned short* __restrict__ cond,
                            unsigned short* __restrict__ acts, int layer) {
  int idx = blockIdx.x * blockDim.x + threadIdx.x;
  if (idx >= TG * NCH) return;
  int c = idx & 255, t = idx >> 8;
  float g0 = xin[(size_t)t * 512 + c]       + bf2f(cond[(size_t)t * CONDCH + layer * 512 + c]);
  float g1 = xin[(size_t)t * 512 + 256 + c] + bf2f(cond[(size_t)t * CONDCH + layer * 512 + 256 + c]);
  float s = 1.0f / (1.0f + expf(-g1));
  acts[(size_t)t * NCH + c] = f2bf(tanhf(g0) * s);
}

__global__ void resskip_kernel(const float* __restrict__ rs,
                               float* __restrict__ xf, unsigned short* __restrict__ xbf,
                               float* __restrict__ skip, int last) {
  int idx = blockIdx.x * blockDim.x + threadIdx.x;
  if (idx >= TG * NCH) return;
  int c = idx & 255, t = idx >> 8;
  if (!last) {
    float nx = xf[(size_t)t * NCH + c] + rs[(size_t)t * 512 + c];
    xf [(size_t)t * NCH + c] = nx;
    xbf[(size_t)t * NCH + c] = f2bf(nx);
    skip[(size_t)t * NCH + c] += rs[(size_t)t * 512 + 256 + c];
  } else {
    skip[(size_t)t * NCH + c] += rs[(size_t)t * NCH + c];
  }
}

__global__ void end_kernel(const float* __restrict__ skip,
                           const float* __restrict__ w, const float* __restrict__ b,
                           float* __restrict__ wnout, int n_c) {
  int idx = blockIdx.x * blockDim.x + threadIdx.x;
  if (idx >= n_c * TG) return;
  int t = idx / n_c, c = idx - t * n_c;
  float acc = b[c];
  #pragma unroll 8
  for (int ci = 0; ci < NCH; ++ci)
    acc += skip[(size_t)t * NCH + ci] * w[c * NCH + ci];
  wnout[(size_t)c * TG + t] = acc;
}

__global__ void coupling_kernel(float* __restrict__ audio, const float* __restrict__ wnout,
                                int off, int n_half) {
  int idx = blockIdx.x * blockDim.x + threadIdx.x;
  if (idx >= n_half * TG) return;
  int c = idx >> 12, t = idx & (TG - 1);
  float* a1 = audio + (size_t)(off + n_half + c) * TG;
  float bias = wnout[(size_t)c * TG + t];
  float logs = wnout[(size_t)(n_half + c) * TG + t];
  a1[t] = (a1[t] - bias) * expf(-logs);
}

__global__ void invert_kernel(const float* __restrict__ W, float* __restrict__ Winv, int n) {
  if (threadIdx.x != 0 || blockIdx.x != 0) return;
  float M[8][16];
  for (int i = 0; i < n; ++i) {
    for (int j = 0; j < n; ++j) { M[i][j] = W[i * n + j]; M[i][n + j] = (i == j) ? 1.f : 0.f; }
  }
  for (int c = 0; c < n; ++c) {
    int p = c; float best = fabsf(M[c][c]);
    for (int r = c + 1; r < n; ++r) { float v = fabsf(M[r][c]); if (v > best) { best = v; p = r; } }
    if (p != c) for (int j = 0; j < 2 * n; ++j) { float tmp = M[c][j]; M[c][j] = M[p][j]; M[p][j] = tmp; }
    float inv = 1.f / M[c][c];
    for (int j = 0; j < 2 * n; ++j) M[c][j] *= inv;
    for (int r = 0; r < n; ++r) {
      if (r == c) continue;
      float f = M[r][c];
      for (int j = 0; j < 2 * n; ++j) M[r][j] -= f * M[c][j];
    }
  }
  for (int i = 0; i < n; ++i)
    for (int j = 0; j < n; ++j) Winv[i * n + j] = M[i][n + j];
}

__global__ void invapply_kernel(float* __restrict__ audio, const float* __restrict__ winv,
                                int off, int n_c) {
  int t = blockIdx.x * blockDim.x + threadIdx.x;
  if (t >= TG) return;
  float in[8], out[8];
  for (int c = 0; c < n_c; ++c) in[c] = audio[(size_t)(off + c) * TG + t];
  for (int c = 0; c < n_c; ++c) {
    float a = 0.f;
    for (int j = 0; j < n_c; ++j) a += winv[c * n_c + j] * in[j];
    out[c] = a;
  }
  for (int c = 0; c < n_c; ++c) audio[(size_t)(off + c) * TG + t] = out[c];
}

__global__ void copyrows_kernel(const float* __restrict__ src, float* __restrict__ audio,
                                int dstRow, int nrows) {
  int idx = blockIdx.x * blockDim.x + threadIdx.x;
  if (idx >= nrows * TG) return;
  audio[(size_t)dstRow * TG + idx] = src[idx];
}

__global__ void final_kernel(const float* __restrict__ audio, float* __restrict__ out) {
  int idx = blockIdx.x * blockDim.x + threadIdx.x;
  if (idx >= T_AUDIO) return;
  int c = idx & 7, tp = idx >> 3;
  out[idx] = audio[(size_t)c * TG + tp];
}

// ---------------------------------------------------------------------------
extern "C" void kernel_launch(void* const* d_in, const int* in_sizes, int n_in,
                              void* d_out, int out_size, void* d_ws, size_t ws_size,
                              hipStream_t stream) {
  (void)in_sizes; (void)n_in; (void)out_size; (void)ws_size;
  const float* spect   = (const float*)d_in[0];
  const float* z_audio = (const float*)d_in[1];
  const float* z_e8    = (const float*)d_in[2];
  const float* z_e4    = (const float*)d_in[3];

  // params flattened as a JAX pytree (dict keys sorted)
  struct Flow {
    const float *cond_b, *cond_w, *end_b, *end_w;
    const float *in_b[8], *in_w[8];
    const float *inv_w;
    const float *rs_b[8], *rs_w[8];
    const float *start_b, *start_w;
  } fl[12];
  int idx = 4;
  for (int k = 0; k < 12; ++k) {
    fl[k].cond_b = (const float*)d_in[idx++];
    fl[k].cond_w = (const float*)d_in[idx++];
    fl[k].end_b  = (const float*)d_in[idx++];
    fl[k].end_w  = (const float*)d_in[idx++];
    for (int i = 0; i < 8; ++i) fl[k].in_b[i] = (const float*)d_in[idx++];
    for (int i = 0; i < 8; ++i) fl[k].in_w[i] = (const float*)d_in[idx++];
    fl[k].inv_w  = (const float*)d_in[idx++];
    for (int i = 0; i < 8; ++i) fl[k].rs_b[i] = (const float*)d_in[idx++];
    for (int i = 0; i < 8; ++i) fl[k].rs_w[i] = (const float*)d_in[idx++];
    fl[k].start_b = (const float*)d_in[idx++];
    fl[k].start_w = (const float*)d_in[idx++];
  }
  const float* up_b = (const float*)d_in[idx++];
  const float* up_w = (const float*)d_in[idx++];

  // ---- workspace carve-out (~82 MB total) ----
  char* wp = (char*)d_ws;
  auto carve = [&](size_t bytes) { char* p = wp; wp += (bytes + 255) & ~(size_t)255; return p; };
  unsigned short* s_bf    = (unsigned short*)carve((size_t)SMEL   * TG * 2);
  unsigned short* cond_bf = (unsigned short*)carve((size_t)CONDCH * TG * 2);
  float*          x_f     = (float*)         carve((size_t)NCH * TG * 4);
  unsigned short* x_bf    = (unsigned short*)carve((size_t)NCH * TG * 2);
  float*          xin_f   = (float*)         carve((size_t)512 * TG * 4);
  unsigned short* acts_bf = (unsigned short*)carve((size_t)NCH * TG * 2);
  float*          rs_f    = (float*)         carve((size_t)512 * TG * 4);
  float*          skip_f  = (float*)         carve((size_t)NCH * TG * 4);
  float*          wnout   = (float*)         carve((size_t)8 * TG * 4);
  float*          audio   = (float*)         carve((size_t)8 * TG * 4);
  float*          winv    = (float*)         carve(64 * 4);
  unsigned short* condw_bf= (unsigned short*)carve((size_t)CONDCH * SMEL * 2);          // 5.2 MB
  unsigned short* inw_bf  = (unsigned short*)carve((size_t)8 * 512 * 768 * 2);          // 6.3 MB
  unsigned short* rsw_bf  = (unsigned short*)carve((size_t)(7 * 512 * 256 + 256 * 256) * 2);
  unsigned short* zeros   = (unsigned short*)carve(256);

  dim3 blk(256);
  (void)hipMemsetAsync(zeros, 0, 256, stream);
  upsample_kernel<<<(NMEL * T_AUDIO + 255) / 256, blk, 0, stream>>>(spect, up_w, up_b, s_bf);
  copyrows_kernel<<<(4 * TG) / 256, blk, 0, stream>>>(z_audio, audio, 4, 4);

  for (int k = 11; k >= 0; --k) {
    const int n_half = (8 - 2 * (k / 4)) / 2;
    const int n_c    = 2 * n_half;
    const int off    = 8 - n_c;
    const Flow& F = fl[k];

    // ---- per-flow weight conversion to bf16 (K made contiguous for conv)
    cvt_w_kernel<<<(CONDCH * SMEL) / 256, blk, 0, stream>>>(F.cond_w, condw_bf, CONDCH * SMEL);
    for (int i = 0; i < 8; ++i)
      cvt_convw_kernel<<<(512 * 768) / 256, blk, 0, stream>>>(F.in_w[i], inw_bf + (size_t)i * 512 * 768);
    for (int i = 0; i < 8; ++i) {
      int n = (i < 7) ? 512 * 256 : 256 * 256;
      cvt_w_kernel<<<n / 256, blk, 0, stream>>>(F.rs_w[i], rsw_bf + (size_t)i * 512 * 256, n);
    }

    // cond 1x1 conv -> bf16 [T][4096]
    gemm_wmma_kernel<SMEL, 1, 1><<<dim3(TG / 128, CONDCH / 128), blk, 0, stream>>>(
        condw_bf, F.cond_b, s_bf, nullptr, cond_bf, CONDCH, 1, zeros);

    start_kernel<<<(NCH * TG) / 256, blk, 0, stream>>>(audio, F.start_w, F.start_b,
                                                       x_f, x_bf, off, n_half);
    (void)hipMemsetAsync(skip_f, 0, (size_t)NCH * TG * 4, stream);

    for (int i = 0; i < 8; ++i) {
      const int dil = 1 << i;
      gemm_wmma_kernel<NCH, 3, 0><<<dim3(TG / 128, 512 / 128), blk, 0, stream>>>(
          inw_bf + (size_t)i * 512 * 768, F.in_b[i], x_bf, xin_f, nullptr, 512, dil, zeros);
      gate_kernel<<<(NCH * TG) / 256, blk, 0, stream>>>(xin_f, cond_bf, acts_bf, i);
      const int Ors = (i < 7) ? 512 : 256;
      gemm_wmma_kernel<NCH, 1, 0><<<dim3(TG / 128, Ors / 128), blk, 0, stream>>>(
          rsw_bf + (size_t)i * 512 * 256, F.rs_b[i], acts_bf, rs_f, nullptr, Ors, 1, zeros);
      resskip_kernel<<<(NCH * TG) / 256, blk, 0, stream>>>(rs_f, x_f, x_bf, skip_f, i == 7);
    }

    end_kernel<<<(n_c * TG + 255) / 256, blk, 0, stream>>>(skip_f, F.end_w, F.end_b, wnout, n_c);
    coupling_kernel<<<(n_half * TG) / 256, blk, 0, stream>>>(audio, wnout, off, n_half);
    invert_kernel<<<1, 32, 0, stream>>>(F.inv_w, winv, n_c);
    invapply_kernel<<<TG / 256, blk, 0, stream>>>(audio, winv, off, n_c);

    if (k % 4 == 0 && k > 0) {
      const float* ze = (k == 8) ? z_e8 : z_e4;
      copyrows_kernel<<<(2 * TG) / 256, blk, 0, stream>>>(ze, audio, off - 2, 2);
    }
  }
  final_kernel<<<T_AUDIO / 256, blk, 0, stream>>>(audio, (float*)d_out);
}
